// RepModule_6725918785954
// MI455X (gfx1250) — matrix operand order repaired and checked by reading, not compiled, not run on hardware
//
#include <hip/hip_runtime.h>

#define N_NODES  100000
#define N_EDGES  1600000
#define ATTR_DIM 16
#define FEAT     64
#define RCUT     5.0f

typedef __attribute__((ext_vector_type(16))) _Float16 v16h;
typedef __attribute__((ext_vector_type(8)))  _Float16 v8h;
typedef __attribute__((ext_vector_type(8)))  float    v8f;

__device__ __forceinline__ v8f wmma16(v16h a, v16h b, v8f c) {
  // D = A(16x32 f16) * B(32x16 f16) + C(16x16 f32)
  return __builtin_amdgcn_wmma_f32_16x16x32_f16(false, a, false, b, (short)0, c,
                                                false, false);
}

// Weights pre-swizzled into B-operand order:
// dst[(((kb*4+c)*32+lane)<<4) + j] = W[kb*32 + (lane>=16?8:0) + (j<8?j:j+8)][c*16 + lane%16]
__device__ __forceinline__ v16h loadB(const _Float16* __restrict__ w, int kb, int c, int lane) {
  return *(const v16h*)(w + (size_t)(((kb * 4 + c) * 32 + lane) << 4));
}

__device__ __forceinline__ void atomAddF(float* p, float v) {
  // Coarse-grained HBM only: permit native global_atomic_add_f32 (no CAS expansion).
  [[clang::atomic(no_remote_memory, no_fine_grained_memory, ignore_denormal_mode)]] {
    __hip_atomic_fetch_add(p, v, __ATOMIC_RELAXED, __HIP_MEMORY_SCOPE_AGENT);
  }
}

// ---------------- weight swizzle (f32 -> f16, B-operand lane layout) ----------------
__global__ __launch_bounds__(256) void swz_kernel(const float* __restrict__ W,
                                                  _Float16* __restrict__ dst,
                                                  int Ksrc, int KB) {
  int t = blockIdx.x * 256 + threadIdx.x;
  int total = KB * 2048;
  if (t >= total) return;
  int j    = t & 15;
  int lane = (t >> 4) & 31;
  int c    = (t >> 9) & 3;
  int kb   = t >> 11;
  int K    = kb * 32 + ((lane & 16) ? 8 : 0) + (j < 8 ? j : j + 8);
  int col  = c * 16 + (lane & 15);
  dst[t] = (K < Ksrc) ? (_Float16)W[K * 64 + col] : (_Float16)0.0f;
}

// ---------------- per-edge geometry: sh[9] + bessel rbf[8] -> f16 ----------------
__global__ __launch_bounds__(256) void edge_geom_kernel(
    const float* __restrict__ pos, const int* __restrict__ ei,
    const float* __restrict__ pvec,
    _Float16* __restrict__ rbf_h, _Float16* __restrict__ sh_h) {
  int e = blockIdx.x * 256 + threadIdx.x;
  if (e >= N_EDGES) return;
  int s = ei[e], d = ei[N_EDGES + e];
  float vx = pos[d * 3 + 0] - pos[s * 3 + 0] + pvec[(size_t)e * 3 + 0];
  float vy = pos[d * 3 + 1] - pos[s * 3 + 1] + pvec[(size_t)e * 3 + 1];
  float vz = pos[d * 3 + 2] - pos[s * 3 + 2] + pvec[(size_t)e * 3 + 2];
  float r   = sqrtf(vx * vx + vy * vy + vz * vz + 1e-12f);
  float inv = 1.0f / r;
  float x = vx * inv, y = vy * inv, z = vz * inv;
  const float s3 = 1.7320508075688772f, s15 = 3.872983346207417f;
  const float s5_4 = 1.118033988749895f, s15_4 = 1.936491673103708f;
  union { _Float16 h[16]; uint4 u[2]; } S;
  S.h[0] = (_Float16)1.0f;
  S.h[1] = (_Float16)(s3 * x);
  S.h[2] = (_Float16)(s3 * y);
  S.h[3] = (_Float16)(s3 * z);
  S.h[4] = (_Float16)(s15 * x * y);
  S.h[5] = (_Float16)(s15 * y * z);
  S.h[6] = (_Float16)(s5_4 * (3.0f * z * z - 1.0f));
  S.h[7] = (_Float16)(s15 * x * z);
  S.h[8] = (_Float16)(s15_4 * (x * x - y * y));
#pragma unroll
  for (int j = 9; j < 16; ++j) S.h[j] = (_Float16)0.0f;
  float rs  = fmaxf(r, 1e-6f);
  float rp  = r / RCUT;
  float rp2 = rp * rp, rp3 = rp2 * rp, rp6 = rp3 * rp3, rp7 = rp6 * rp, rp8 = rp7 * rp;
  float env  = (rp < 1.0f) ? (1.0f - 28.0f * rp6 + 48.0f * rp7 - 21.0f * rp8) : 0.0f;
  float pref = sqrtf(2.0f / RCUT) / rs * env;
  float k0   = 3.14159265358979323846f * rs / RCUT;
  union { _Float16 h[8]; uint4 u; } R;
#pragma unroll
  for (int n = 1; n <= 8; ++n) R.h[n - 1] = (_Float16)(pref * __sinf((float)n * k0));
  uint4* sp = (uint4*)(sh_h + (size_t)e * 16);
  sp[0] = S.u[0];
  sp[1] = S.u[1];
  *(uint4*)(rbf_h + (size_t)e * 8) = R.u;
}

// ---------------- embedding: h0 = elem_table[x] @ W_embed (K=16 pad 32) ----------------
__global__ __launch_bounds__(256) void embed_kernel(
    const int* __restrict__ x, const float* __restrict__ elem_table,
    const _Float16* __restrict__ Wembs, float* __restrict__ h) {
  int wave = threadIdx.x >> 5, lane = threadIdx.x & 31;
  int tile = blockIdx.x * 8 + wave;
  if (tile >= N_NODES / 16) return;
  int half = lane & 15;
  int K0   = (lane >= 16) ? 8 : 0;
  int moff = K0;
  int xid  = x[tile * 16 + half];
  const float* arow = elem_table + xid * ATTR_DIM;
  v16h aA = {};
#pragma unroll
  for (int j = 0; j < 8; ++j) aA[j] = (_Float16)arow[K0 + j];
#pragma unroll
  for (int c = 0; c < 4; ++c) {
    v8f acc = {};
    acc = wmma16(aA, loadB(Wembs, 0, c, lane), acc);
    int col = half + 16 * c;
#pragma unroll
    for (int i = 0; i < 8; ++i)
      h[(size_t)(tile * 16 + i + moff) * FEAT + col] = acc[i];
  }
}

// ---------------- node linears: Hm = h@Wmsg ; Hs = h@Wself + attr@Wattr ; agg=0 --------
__global__ __launch_bounds__(256) void node_linear_kernel(
    const float* __restrict__ h, const int* __restrict__ x,
    const float* __restrict__ elem_table,
    const _Float16* __restrict__ Wmsgs, const _Float16* __restrict__ Wselfs,
    const _Float16* __restrict__ Wattrs,
    float* __restrict__ Hm, float* __restrict__ Hs, float* __restrict__ agg) {
  int wave = threadIdx.x >> 5, lane = threadIdx.x & 31;
  int tile = blockIdx.x * 8 + wave;
  if (tile >= N_NODES / 16) return;
  int half = lane & 15;
  int K0   = (lane >= 16) ? 8 : 0;
  int moff = K0;
  const float* hrow = h + (size_t)(tile * 16 + half) * FEAT;
  v16h a0 = {}, a1 = {}, aA = {};
#pragma unroll
  for (int j = 0; j < 8; ++j) {
    a0[j]     = (_Float16)hrow[K0 + j];
    a0[j + 8] = (_Float16)hrow[K0 + 16 + j];
    a1[j]     = (_Float16)hrow[32 + K0 + j];
    a1[j + 8] = (_Float16)hrow[32 + K0 + 16 + j];
  }
  int xid = x[tile * 16 + half];
  const float* arow = elem_table + xid * ATTR_DIM;
#pragma unroll
  for (int j = 0; j < 8; ++j) aA[j] = (_Float16)arow[K0 + j];
#pragma unroll
  for (int c = 0; c < 4; ++c) {
    v8f m = {}, s = {};
    m = wmma16(a0, loadB(Wmsgs, 0, c, lane), m);
    m = wmma16(a1, loadB(Wmsgs, 1, c, lane), m);
    s = wmma16(a0, loadB(Wselfs, 0, c, lane), s);
    s = wmma16(a1, loadB(Wselfs, 1, c, lane), s);
    s = wmma16(aA, loadB(Wattrs, 0, c, lane), s);
    int col = half + 16 * c;
#pragma unroll
    for (int i = 0; i < 8; ++i) {
      size_t o = (size_t)(tile * 16 + i + moff) * FEAT + col;
      Hm[o] = m[i];
      Hs[o] = s[i];
      agg[o] = 0.0f;
    }
  }
}

// ---------------- edge path: radial MLP + sh mix (WMMA), gather Hm, scatter-add -------
__global__ __launch_bounds__(256) void edge_msg_kernel(
    const _Float16* __restrict__ rbf_h, const _Float16* __restrict__ sh_h,
    const int* __restrict__ ei, const float* __restrict__ Hm,
    const _Float16* __restrict__ Wr1s, const _Float16* __restrict__ Wr2s,
    const _Float16* __restrict__ Wshs, float* __restrict__ agg) {
  __shared__ _Float16 lds_hidden[8][16][72];  // [wave][M][K], stride 72 avoids bank conflicts
  int wave = threadIdx.x >> 5, lane = threadIdx.x & 31;
  int tile = blockIdx.x * 8 + wave;           // 1.6M/16 tiles, divisible by 8
  size_t e0 = (size_t)tile * 16;
  int half = lane & 15;
  int K0   = (lane >= 16) ? 8 : 0;
  int moff = K0;

  v16h a_rbf = {}, a_sh = {};
  if (lane < 16) {
    v8h r8 = *(const v8h*)(rbf_h + (e0 + lane) * 8);
    v8h s8 = *(const v8h*)(sh_h + (e0 + lane) * 16);
#pragma unroll
    for (int j = 0; j < 8; ++j) { a_rbf[j] = r8[j]; a_sh[j] = s8[j]; }
  } else {
    a_sh[0] = sh_h[(e0 + (lane - 16)) * 16 + 8];  // K=8 lives in upper lane group
  }
  int mySrc = ei[e0 + half];
  int myDst = ei[(size_t)N_EDGES + e0 + half];

  // hidden = silu(rbf @ W_r1)  (K=8 padded to 32) -> LDS as f16
#pragma unroll
  for (int c = 0; c < 4; ++c) {
    v8f acc = {};
    acc = wmma16(a_rbf, loadB(Wr1s, 0, c, lane), acc);
    int nfix = half + 16 * c;
#pragma unroll
    for (int i = 0; i < 8; ++i) {
      float v = acc[i];
      lds_hidden[wave][i + moff][nfix] = (_Float16)(v / (1.0f + __expf(-v)));
    }
  }
  __syncthreads();

  // re-gather hidden in A-operand layout (K=64 -> two K-blocks)
  v16h ah0 = {}, ah1 = {};
  {
    const _Float16* hr = &lds_hidden[wave][half][0];
#pragma unroll
    for (int j = 0; j < 8; ++j) {
      ah0[j]     = hr[K0 + j];
      ah0[j + 8] = hr[K0 + 16 + j];
      ah1[j]     = hr[32 + K0 + j];
      ah1[j + 8] = hr[32 + K0 + 16 + j];
    }
  }
#pragma unroll
  for (int c = 0; c < 4; ++c) {
    v8f radial = {};
    radial = wmma16(ah0, loadB(Wr2s, 0, c, lane), radial);
    radial = wmma16(ah1, loadB(Wr2s, 1, c, lane), radial);
    v8f shm = {};
    shm = wmma16(a_sh, loadB(Wshs, 0, c, lane), shm);
    int col = half + 16 * c;
#pragma unroll
    for (int i = 0; i < 8; ++i) {
      int m = i + moff;
      int srcRow = __shfl(mySrc, m, 32);
      int dstRow = __shfl(myDst, m, 32);
      float msg = Hm[(size_t)srcRow * FEAT + col] * radial[i] * shm[i];
      atomAddF(&agg[(size_t)dstRow * FEAT + col], msg);
    }
  }
}

// ---------------- h_next = silu(Hs + agg) ----------------
__global__ __launch_bounds__(256) void finalize_kernel(const float* __restrict__ Hs,
                                                       const float* __restrict__ agg,
                                                       float* __restrict__ out, int n) {
  int i = blockIdx.x * 256 + threadIdx.x;
  if (i < n) {
    float v = Hs[i] + agg[i];
    out[i] = v / (1.0f + __expf(-v));
  }
}

extern "C" void kernel_launch(void* const* d_in, const int* in_sizes, int n_in,
                              void* d_out, int out_size, void* d_ws, size_t ws_size,
                              hipStream_t stream) {
  const int*   x     = (const int*)d_in[0];
  const float* pos   = (const float*)d_in[1];
  const int*   ei    = (const int*)d_in[2];
  const float* pvec  = (const float*)d_in[3];
  const float* elem  = (const float*)d_in[5];
  const float* Wemb  = (const float*)d_in[6];
  const float* Wattr = (const float*)d_in[7];
  const float* Wself = (const float*)d_in[8];
  const float* Wmsg  = (const float*)d_in[9];
  const float* Wr1   = (const float*)d_in[10];
  const float* Wr2   = (const float*)d_in[11];
  const float* Wsh   = (const float*)d_in[12];

  size_t off = 0;
  auto alloc = [&](size_t bytes) -> void* {
    off = (off + 255) & ~(size_t)255;
    void* p = (char*)d_ws + off;
    off += bytes;
    return p;
  };
  _Float16* rbf_h = (_Float16*)alloc((size_t)N_EDGES * 8 * 2);
  _Float16* sh_h  = (_Float16*)alloc((size_t)N_EDGES * 16 * 2);
  float* hA  = (float*)alloc((size_t)N_NODES * FEAT * 4);
  float* hB  = (float*)alloc((size_t)N_NODES * FEAT * 4);
  float* Hm  = (float*)alloc((size_t)N_NODES * FEAT * 4);
  float* Hs  = (float*)alloc((size_t)N_NODES * FEAT * 4);
  float* agg = (float*)alloc((size_t)N_NODES * FEAT * 4);
  _Float16* wswz = (_Float16*)alloc((size_t)(2048 + 3 * 18432) * 2);

  // swizzle all weights once per call (tiny)
  _Float16* Wemb_s = wswz;
  swz_kernel<<<8, 256, 0, stream>>>(Wemb, Wemb_s, 16, 1);
  for (int l = 0; l < 3; ++l) {
    _Float16* base = wswz + 2048 + (size_t)l * 18432;
    swz_kernel<<<8,  256, 0, stream>>>(Wattr + (size_t)l * 16 * 64, base,          16, 1);
    swz_kernel<<<16, 256, 0, stream>>>(Wself + (size_t)l * 64 * 64, base + 2048,   64, 2);
    swz_kernel<<<16, 256, 0, stream>>>(Wmsg  + (size_t)l * 64 * 64, base + 6144,   64, 2);
    swz_kernel<<<8,  256, 0, stream>>>(Wr1   + (size_t)l * 8  * 64, base + 10240,   8, 1);
    swz_kernel<<<16, 256, 0, stream>>>(Wr2   + (size_t)l * 64 * 64, base + 12288,  64, 2);
    swz_kernel<<<8,  256, 0, stream>>>(Wsh   + (size_t)l * 9  * 64, base + 16384,   9, 1);
  }

  edge_geom_kernel<<<(N_EDGES + 255) / 256, 256, 0, stream>>>(pos, ei, pvec, rbf_h, sh_h);

  int nodeBlocks = (N_NODES / 16 + 7) / 8;
  embed_kernel<<<nodeBlocks, 256, 0, stream>>>(x, elem, Wemb_s, hA);

  float* hcur = hA;
  float* hnext = hB;
  for (int l = 0; l < 3; ++l) {
    _Float16* base = wswz + 2048 + (size_t)l * 18432;
    node_linear_kernel<<<nodeBlocks, 256, 0, stream>>>(
        hcur, x, elem, base + 6144 /*msg*/, base + 2048 /*self*/, base /*attr*/,
        Hm, Hs, agg);
    edge_msg_kernel<<<N_EDGES / 16 / 8, 256, 0, stream>>>(
        rbf_h, sh_h, ei, Hm, base + 10240 /*r1*/, base + 12288 /*r2*/,
        base + 16384 /*sh*/, agg);
    float* outp = (l == 2) ? (float*)d_out : hnext;
    finalize_kernel<<<(N_NODES * FEAT + 255) / 256, 256, 0, stream>>>(
        Hs, agg, outp, N_NODES * FEAT);
    float* t = hcur; hcur = hnext; hnext = t;
  }
}
